// PositiveInnerUMNN_35734127903347
// MI455X (gfx1250) — matrix operand order repaired and checked by reading, not compiled, not run on hardware
//
#include <hip/hip_runtime.h>
#include <hip/hip_bf16.h>

typedef __attribute__((ext_vector_type(16))) _Float16 v16h;
typedef __attribute__((ext_vector_type(8)))  _Float16 v8h;
typedef __attribute__((ext_vector_type(8)))  float    v8f;

#define B_TOTAL 16384
#define KSTEPS  51
#define IN_D    64
#define HD      128
#define WAVES   4
#define BLOCK   (WAVES * 32)
#define BT      (WAVES * 16)     // 64 batch rows per block
#define AST     144              // act row stride in halves (128 + 16 pad, 16B aligned)

// ---------------- setup: xmax reduction + Clenshaw-Curtis weights/nodes ----------------
// ws layout (floats): [0]=xmax(+10), [8..58]=cc_weights, [64..114]=steps
__global__ __launch_bounds__(256) void umnn_setup_kernel(const float* __restrict__ x,
                                                         float* __restrict__ ws) {
  __shared__ float red[256];
  const int tid = threadIdx.x;
  float mx = -3.4e38f;
  for (int i = tid; i < B_TOTAL; i += 256) mx = fmaxf(mx, x[i]);
  red[tid] = mx;
  __syncthreads();
  for (int s = 128; s > 0; s >>= 1) {
    if (tid < s) red[tid] = fmaxf(red[tid], red[tid + s]);
    __syncthreads();
  }
  if (tid == 0) ws[0] = red[0] + 10.0f;
  if (tid < KSTEPS) {
    const double PI = 3.14159265358979323846;
    const int i = tid;
    double acc = 0.0;
    for (int j = 0; j <= 50; ++j) {
      double Wj = (j == 0) ? 1.0 : ((j & 1) ? 0.0 : 2.0 / (1.0 - (double)(j * j)));
      double lv;
      if (i == 0)       lv = 0.5;
      else if (i == 50) lv = 0.5 * cos((double)j * PI);
      else              lv = cos((double)i * (double)j * PI / 50.0);
      acc += Wj * lv;
    }
    ws[8 + i]  = (float)(acc * 2.0 / 50.0);
    ws[64 + i] = (float)cos((double)i * PI / 50.0);
  }
}

// ---------------- main kernel: batched MLP via f16 WMMA, k-loop over CC nodes ----------------
// Key structure: layer-1 is split as y1 = ReLU(X*W1[:,0] + g), g = W1[:,1:]@h + b1 computed
// ONCE per tile with WMMA (k-invariant); the k-loop runs only layers 2/3 on the matrix pipe.
__global__ __launch_bounds__(BLOCK) void umnn_main_kernel(
    const float* __restrict__ x,  const float* __restrict__ h,
    const float* __restrict__ W1, const float* __restrict__ b1,
    const float* __restrict__ W2, const float* __restrict__ b2,
    const float* __restrict__ W3, const float* __restrict__ b3,
    const float* __restrict__ W4, const float* __restrict__ b4,
    const float* __restrict__ ws, float* __restrict__ out)
{
  // B fragments pre-swizzled: [frag][lane][16 halves], frag = ktile*8 + ntile
  __shared__ __align__(32) _Float16 sB1[16 * 32 * 16];       // 16 KB (init only)
  __shared__ __align__(32) _Float16 sB2[32 * 32 * 16];       // 32 KB
  __shared__ __align__(32) _Float16 sB3[32 * 32 * 16];       // 32 KB
  __shared__ __align__(32) _Float16 sAct[WAVES * 16 * AST];  // 18 KB per-wave scratch
  __shared__ float sStep[KSTEPS], sCCW[KSTEPS];

  const int tid     = threadIdx.x;
  const int lane    = tid & 31;
  const int wave    = tid >> 5;
  const int m       = lane & 15;        // row-within-tile (A) / column (B,C)
  const int halfsel = (lane >> 4) & 1;  // lane half

  // ---- cooperative fill: weights -> f16 B-fragment layout in LDS ----
  // B (KxN) 16-bit layout: elem j of lane half hs <-> K = 8*hs + (j<8 ? j : j+8).
  for (int e = tid; e < HD * IN_D; e += BLOCK) {
    int n = e >> 6, k = e & 63;
    int k2 = k & 31;
    int j  = (k2 & 7) + ((k2 >> 1) & 8);
    int ln = (n & 15) + ((k2 & 8) << 1);
    int fr = ((k >> 5) << 3) + (n >> 4);
    sB1[(fr * 32 + ln) * 16 + j] = (_Float16)W1[n * IN_D + k];
  }
  for (int e = tid; e < HD * HD; e += BLOCK) {
    int n = e >> 7, k = e & 127;
    int k2 = k & 31;
    int j  = (k2 & 7) + ((k2 >> 1) & 8);
    int ln = (n & 15) + ((k2 & 8) << 1);
    int fr = ((k >> 5) << 3) + (n >> 4);
    int idx = (fr * 32 + ln) * 16 + j;
    sB2[idx] = (_Float16)W2[n * HD + k];
    sB3[idx] = (_Float16)W3[n * HD + k];
  }
  for (int i = tid; i < KSTEPS; i += BLOCK) { sStep[i] = ws[64 + i]; sCCW[i] = ws[8 + i]; }
  __syncthreads();

  // ---- per-lane constants ----
  float bias2[8], bias3[8], w4r[8], w1c0[8];
#pragma unroll
  for (int t = 0; t < 8; ++t) {
    bias2[t] = b2[t * 16 + m];
    bias3[t] = b3[t * 16 + m];
    w4r[t]   = W4[t * 16 + m];
    w1c0[t]  = W1[(t * 16 + m) * IN_D];   // W1[:,0] column (X weight), col = t*16+m
  }
  const float b4s  = b4[0];
  const float xmax = ws[0];

  // ---- batch rows for this wave ----
  const int bbase = blockIdx.x * BT + wave * 16;
  const int brow  = bbase + m;
  // x0 per C-layout row: row v + 8*halfsel
  float x0r[8];
#pragma unroll
  for (int v = 0; v < 8; ++v) x0r[v] = x[bbase + 8 * halfsel + v];

  // ---- layer-1 A fragments with X column zeroed (h part only) ----
  // A (MxK) 16-bit layout: row M = lane%16; elem j <-> K = 8*halfsel + (j<8 ? j : j+8)
  const float* hrow = h + (long)brow * (IN_D - 1);
  v16h a10, a11;
#pragma unroll
  for (int j = 0; j < 16; ++j) {
    int kk = (halfsel << 3) + ((j < 8) ? j : (j + 8));
    a10[j] = (kk == 0) ? (_Float16)0.0f : (_Float16)hrow[kk - 1];  // K=0 (X col) = 0
    a11[j] = (_Float16)hrow[31 + kk];                              // cols 32..63 -> h[31..62]
  }

  // ---- g = W1[:,1:]@h + b1, computed once, kept resident in C-fragment layout ----
  v8f g[8];
#pragma unroll
  for (int t = 0; t < 8; ++t) {
    v8f acc = {};
    v16h bf0 = *(const v16h*)(sB1 + ((0 * 8 + t) * 32 + lane) * 16);
    acc = __builtin_amdgcn_wmma_f32_16x16x32_f16(false, a10, false, bf0, (short)0, acc, false, false);
    v16h bf1 = *(const v16h*)(sB1 + ((1 * 8 + t) * 32 + lane) * 16);
    acc = __builtin_amdgcn_wmma_f32_16x16x32_f16(false, a11, false, bf1, (short)0, acc, false, false);
    const float bb = b1[t * 16 + m];
#pragma unroll
    for (int v = 0; v < 8; ++v) g[t][v] = acc[v] + bb;
  }

  _Float16* act = sAct + wave * 16 * AST;
  float zacc[8];
#pragma unroll
  for (int v = 0; v < 8; ++v) zacc[v] = 0.0f;

  // ---- quadrature loop ----
  for (int k = 0; k < KSTEPS; ++k) {
    const float sfac = (sStep[k] + 1.0f) * 0.5f;
    float Xr[8];
#pragma unroll
    for (int v = 0; v < 8; ++v) Xr[v] = fmaf(xmax - x0r[v], sfac, x0r[v]);

    // ---- layer 1 (VALU only): relu(X*w1c0 + g) -> act (row-major f16) ----
#pragma unroll
    for (int t = 0; t < 8; ++t) {
#pragma unroll
      for (int v = 0; v < 8; ++v) {
        float y = fmaf(Xr[v], w1c0[t], g[t][v]);
        y = y > 0.0f ? y : 0.0f;
        act[(v + 8 * halfsel) * AST + t * 16 + m] = (_Float16)y;
      }
    }
    __asm__ volatile("" ::: "memory");

    // ---- reload as layer-2 A fragments (DS in-order within wave) ----
    v16h a2[4];
#pragma unroll
    for (int s = 0; s < 4; ++s) {
      const _Float16* p = act + m * AST + s * 32 + (halfsel << 3);
      v8h lo = *(const v8h*)p;
      v8h hi = *(const v8h*)(p + 16);
      a2[s] = __builtin_shufflevector(lo, hi, 0,1,2,3,4,5,6,7,8,9,10,11,12,13,14,15);
    }
    __asm__ volatile("" ::: "memory");

    // ---- layer 2: [16x128] x [128x128] ----
#pragma unroll
    for (int t = 0; t < 8; ++t) {
      v8f acc = {};
#pragma unroll
      for (int s = 0; s < 4; ++s) {
        v16h bf = *(const v16h*)(sB2 + ((s * 8 + t) * 32 + lane) * 16);
        acc = __builtin_amdgcn_wmma_f32_16x16x32_f16(false, a2[s], false, bf, (short)0, acc, false, false);
      }
#pragma unroll
      for (int v = 0; v < 8; ++v) {
        float y = acc[v] + bias2[t];
        y = y > 0.0f ? y : 0.0f;
        act[(v + 8 * halfsel) * AST + t * 16 + m] = (_Float16)y;
      }
    }
    __asm__ volatile("" ::: "memory");

    v16h a3[4];
#pragma unroll
    for (int s = 0; s < 4; ++s) {
      const _Float16* p = act + m * AST + s * 32 + (halfsel << 3);
      v8h lo = *(const v8h*)p;
      v8h hi = *(const v8h*)(p + 16);
      a3[s] = __builtin_shufflevector(lo, hi, 0,1,2,3,4,5,6,7,8,9,10,11,12,13,14,15);
    }
    __asm__ volatile("" ::: "memory");

    // ---- layer 3 + fused layer 4 (128->1) epilogue ----
    float part[8];
#pragma unroll
    for (int v = 0; v < 8; ++v) part[v] = 0.0f;
#pragma unroll
    for (int t = 0; t < 8; ++t) {
      v8f acc = {};
#pragma unroll
      for (int s = 0; s < 4; ++s) {
        v16h bf = *(const v16h*)(sB3 + ((s * 8 + t) * 32 + lane) * 16);
        acc = __builtin_amdgcn_wmma_f32_16x16x32_f16(false, a3[s], false, bf, (short)0, acc, false, false);
      }
#pragma unroll
      for (int v = 0; v < 8; ++v) {
        float y = acc[v] + bias3[t];
        y = y > 0.0f ? y : 0.0f;
        part[v] = fmaf(y, w4r[t], part[v]);  // dot with W4 column slice
      }
    }
    // cross-lane reduce over the 16 lanes of each half (rows v / v+8)
    const float cw = sCCW[k];
#pragma unroll
    for (int v = 0; v < 8; ++v) {
      float p = part[v];
      p += __shfl_xor(p, 1, 32);
      p += __shfl_xor(p, 2, 32);
      p += __shfl_xor(p, 4, 32);
      p += __shfl_xor(p, 8, 32);
      float y = p + b4s;
      // elu(y)+1 == (y>0 ? y+1 : exp(y))
      float f = (y > 0.0f) ? (y + 1.0f) : __expf(y);
      zacc[v] = fmaf(f, cw, zacc[v]);
    }
  }

  // ---- finalize: z * (xmax - x0) * 0.5 ; one lane per half writes its 8 rows ----
  if (m == 0) {
#pragma unroll
    for (int v = 0; v < 8; ++v) {
      int r = bbase + 8 * halfsel + v;
      out[r] = zacc[v] * (xmax - x0r[v]) * 0.5f;
    }
  }
}

extern "C" void kernel_launch(void* const* d_in, const int* in_sizes, int n_in,
                              void* d_out, int out_size, void* d_ws, size_t ws_size,
                              hipStream_t stream) {
  (void)in_sizes; (void)n_in; (void)out_size; (void)ws_size;
  const float* x  = (const float*)d_in[0];
  const float* h  = (const float*)d_in[1];
  const float* W1 = (const float*)d_in[2];
  const float* b1 = (const float*)d_in[3];
  const float* W2 = (const float*)d_in[4];
  const float* b2 = (const float*)d_in[5];
  const float* W3 = (const float*)d_in[6];
  const float* b3 = (const float*)d_in[7];
  const float* W4 = (const float*)d_in[8];
  const float* b4 = (const float*)d_in[9];
  float* ws  = (float*)d_ws;
  float* out = (float*)d_out;

  umnn_setup_kernel<<<dim3(1), dim3(256), 0, stream>>>(x, ws);
  umnn_main_kernel<<<dim3(B_TOTAL / BT), dim3(BLOCK), 0, stream>>>(
      x, h, W1, b1, W2, b2, W3, b3, W4, b4, ws, out);
}
